// MaskedMultiHeadAttention_59107339927724
// MI455X (gfx1250) — matrix operand order repaired and checked by reading, not compiled, not run on hardware
//
#include <hip/hip_runtime.h>
#include <hip/hip_bf16.h>
#include <stdint.h>

#define BATCH 4
#define SEQ   2048
#define EMB   1024
#define NH    16
#define DH    64

#define BM 128          // GEMM macro-tile M
#define BN 128          // GEMM macro-tile N
#define BK 32           // GEMM k-slab (= WMMA K)
#define LST 40          // LDS row stride in bf16 (80B: conflict-free period 16)

typedef __attribute__((ext_vector_type(8)))  float  v8f;
typedef __attribute__((ext_vector_type(16))) __bf16 v16bf;
typedef __attribute__((ext_vector_type(8)))  __bf16 v8bf;

static __device__ __forceinline__ v16bf cat8(v8bf lo, v8bf hi) {
  v16bf r;
#pragma unroll
  for (int i = 0; i < 8; ++i) { r[i] = lo[i]; r[8 + i] = hi[i]; }
  return r;
}

static __device__ __forceinline__ v8bf cvt8(float4 a, float4 b) {
  v8bf r;
  r[0]=(__bf16)a.x; r[1]=(__bf16)a.y; r[2]=(__bf16)a.z; r[3]=(__bf16)a.w;
  r[4]=(__bf16)b.x; r[5]=(__bf16)b.y; r[6]=(__bf16)b.z; r[7]=(__bf16)b.w;
  return r;
}

// ---------------------------------------------------------------------------
// LDS-tiled, DOUBLE-BUFFERED GEMM: Out = A(MxK,f32) @ W(KxN,f32) + bias(N).
// 256-thread block -> 128x128 macro-tile; 8 waves in 2(M)x4(N) layout, each
// wave owns a 64x32 sub-tile = 4x2 WMMA accumulators.  Slab k+1 is staged
// into the ping LDS buffer while the 8 WMMAs consume the pong buffer; one
// barrier per k-iteration.
// mode 0: bf16 out [B][H][T][DH] (Q,K) | mode 1: bf16 [B][H][DH][T] (V)
// mode 2: f32 out row-major MxN (final projection)
// ---------------------------------------------------------------------------
__global__ __launch_bounds__(256)
void gemm_bf16_wmma(const float* __restrict__ A, const float* __restrict__ W,
                    const float* __restrict__ bias, void* __restrict__ outp,
                    int M, int N, int K, int mode)
{
  __shared__ __bf16 Alds[2][BM * LST];   // [m][k]  (row-major, padded)
  __shared__ __bf16 Wlds[2][BN * LST];   // [n][k]  (W transposed, padded)

  const int tid  = threadIdx.x;
  const int lane = tid & 31;
  const int wv   = tid >> 5;          // 0..7
  const int wm   = wv >> 2;           // 0..1 : M offset wm*64
  const int wn   = wv & 3;            // 0..3 : N offset wn*32
  const int half = lane >> 4;
  const int l16  = lane & 15;
  const int koffA = half ? 8 : 0;     // A-layout k-chunk base per lane half

  const int nbn = N / BN;
  const int bm  = blockIdx.x / nbn;
  const int bn  = blockIdx.x % nbn;

  // cooperative loader mapping
  const int arow = tid >> 1;              // 0..127 (A row in tile)
  const int acol = (tid & 1) * 16;        // 0 or 16 (k chunk)
  const int wrow = tid >> 3;              // 0..31  (k row of W)
  const int wcol = (tid & 7) * 16;        // n chunk base

  const float* agb = A + (size_t)(bm * BM + arow) * K + acol;
  const float* wgb = W + (size_t)wrow * N + bn * BN + wcol;

  auto stage = [&](int buf, int kk) {
    // A tile: coalesced f32 read -> bf16 LDS rows
    const float* ag = agb + kk;
    float4 a0 = *(const float4*)(ag + 0);
    float4 a1 = *(const float4*)(ag + 4);
    float4 a2 = *(const float4*)(ag + 8);
    float4 a3 = *(const float4*)(ag + 12);
    *(v8bf*)&Alds[buf][arow * LST + acol + 0] = cvt8(a0, a1);
    *(v8bf*)&Alds[buf][arow * LST + acol + 8] = cvt8(a2, a3);
    // W tile: coalesced f32 read, transposed bf16 LDS write
    const float* wg = wgb + (size_t)kk * N;
    float4 w0 = *(const float4*)(wg + 0);
    float4 w1 = *(const float4*)(wg + 4);
    float4 w2 = *(const float4*)(wg + 8);
    float4 w3 = *(const float4*)(wg + 12);
    float wf[16] = {w0.x,w0.y,w0.z,w0.w, w1.x,w1.y,w1.z,w1.w,
                    w2.x,w2.y,w2.z,w2.w, w3.x,w3.y,w3.z,w3.w};
#pragma unroll
    for (int j = 0; j < 16; ++j)
      Wlds[buf][(wcol + j) * LST + wrow] = (__bf16)wf[j];
  };

  v8f acc[4][2] = {};

  stage(0, 0);
  __syncthreads();

  int buf = 0;
  for (int kk = 0; kk < K; kk += BK) {
    // prefetch slab kk+2*BK while staging kk+BK (speculative, no counters)
    if (kk + 2 * BK < K) {
      __builtin_prefetch(agb + kk + 2 * BK, 0, 1);
      __builtin_prefetch(wgb + (size_t)(kk + 2 * BK) * N, 0, 1);
    }
    // stage next slab into the ping buffer (overlaps with WMMAs below)
    if (kk + BK < K) stage(buf ^ 1, kk + BK);

    // ---- B operands (32x16, B-layout): k = half*8 + 0..7, 16 + half*8 + 0..7 ----
    v16bf bop[2];
#pragma unroll
    for (int j = 0; j < 2; ++j) {
      const __bf16* p = &Wlds[buf][(wn * 32 + j * 16 + l16) * LST + half * 8];
      bop[j] = cat8(*(const v8bf*)p, *(const v8bf*)(p + 16));
    }
    // ---- A operands + 8 WMMAs ----
#pragma unroll
    for (int i = 0; i < 4; ++i) {
      const __bf16* p = &Alds[buf][(wm * 64 + i * 16 + l16) * LST + koffA];
      v16bf aop = cat8(*(const v8bf*)p, *(const v8bf*)(p + 16));
#pragma unroll
      for (int j = 0; j < 2; ++j)
        acc[i][j] = __builtin_amdgcn_wmma_f32_16x16x32_bf16(
            false, aop, false, bop[j], (short)0, acc[i][j], false, false);
    }
    __syncthreads();       // compute on buf done + stores to buf^1 visible
    buf ^= 1;
  }

  // ---- epilogue: bias + store per 16x16 tile (D-layout) ----
#pragma unroll
  for (int i = 0; i < 4; ++i) {
#pragma unroll
    for (int j = 0; j < 2; ++j) {
      const int n = bn * BN + wn * 32 + j * 16 + l16;
      const float bnv = bias ? bias[n] : 0.f;
#pragma unroll
      for (int r = 0; r < 8; ++r) {
        const int mm = bm * BM + wm * 64 + i * 16 + (half ? 8 + r : r);
        const float val = acc[i][j][r] + bnv;
        if (mode == 2) {
          ((float*)outp)[(size_t)mm * N + n] = val;
        } else {
          const int b = mm >> 11, t = mm & (SEQ - 1);   // SEQ = 2048
          const int h = n >> 6,  d = n & 63;
          __bf16* o = (__bf16*)outp;
          const size_t idx = (mode == 0)
            ? ((((size_t)b * NH + h) * SEQ + t) * DH + d)
            : ((((size_t)b * NH + h) * DH + d) * SEQ + t);
          o[idx] = (__bf16)val;
        }
      }
    }
  }
}

// ---------------------------------------------------------------------------
// Flash attention, one wave per (b, h, 16-query tile).  Scores computed
// TRANSPOSED (S^T = K * Q^T) so each lane owns one query column: softmax
// stats are in-register + one lane-half shuffle; O^T accumulators (4 tiles
// of 16d x 16q) live in registers across the whole key loop.
// ---------------------------------------------------------------------------
__global__ __launch_bounds__(256)
void attn_wmma(const __bf16* __restrict__ Q, const __bf16* __restrict__ Km,
               const __bf16* __restrict__ Vt, const unsigned char* __restrict__ pad,
               float* __restrict__ O)
{
  const int lane = threadIdx.x & 31;
  const int wave = (int)((blockIdx.x * blockDim.x + threadIdx.x) >> 5);
  const int qtiles = SEQ / 16;
  const int qt = wave % qtiles;
  const int h  = (wave / qtiles) % NH;
  const int b  = wave / (qtiles * NH);
  const int qi = qt << 4;
  const int half = lane >> 4;
  const int l16  = lane & 15;
  const int qg = qi + l16;                 // query column owned by this lane
  const size_t bh = (size_t)b * NH + h;

  // Q operand (B-layout, d is the WMMA K-dim): chunk c covers d = c*32..+31
  v16bf qop[2];
#pragma unroll
  for (int c = 0; c < 2; ++c) {
    const __bf16* p = Q + (bh * SEQ + qg) * DH + c * 32 + half * 8;
    qop[c] = cat8(*(const v8bf*)p, *(const v8bf*)(p + 16));
  }

  float mrun = -1e30f, lrun = 0.f;
  v8f oacc[4] = {{}, {}, {}, {}};
  const int koff = half ? 8 : 0;
  const int nkb = (qi + 15) / 32 + 1;      // causal: keys up to qi+15

  for (int kb = 0; kb < nkb; ++kb) {
    const int kj = kb * 32;

    // speculative prefetch of next key block (global_prefetch_b8)
    if (kb + 1 < nkb) {
      const int kjn = kj + 32;
      __builtin_prefetch(Km + (bh * SEQ + kjn + lane) * DH, 0, 1);
      __builtin_prefetch(Vt + (bh * DH + lane) * SEQ + kjn, 0, 1);
      __builtin_prefetch(Vt + (bh * DH + 32 + lane) * SEQ + kjn, 0, 1);
    }

    // ---- S^T tiles: 2 sub-blocks of 16 keys x 16 queries ----
    v8f sacc[2];
#pragma unroll
    for (int s = 0; s < 2; ++s) {
      v8f z = {};
#pragma unroll
      for (int c = 0; c < 2; ++c) {
        const __bf16* kp = Km + (bh * SEQ + (kj + s * 16 + l16)) * DH + c * 32 + koff;
        v16bf kop = cat8(*(const v8bf*)kp, *(const v8bf*)(kp + 16));
        z = __builtin_amdgcn_wmma_f32_16x16x32_bf16(false, kop, false, qop[c],
                                                    (short)0, z, false, false);
      }
      sacc[s] = z;
    }

    // ---- masking + online-softmax stats (per lane = per query column) ----
    float pv[2][8];
    float bmax = -INFINITY;
#pragma unroll
    for (int s = 0; s < 2; ++s)
#pragma unroll
      for (int r = 0; r < 8; ++r) {
        const int kg = kj + s * 16 + (half ? 8 + r : r);
        const bool ok = (kg <= qg) && (pad[(size_t)b * SEQ + kg] != 0);
        const float sv = ok ? sacc[s][r] * 0.125f : -INFINITY;
        pv[s][r] = sv;
        bmax = fmaxf(bmax, sv);
      }
    bmax = fmaxf(bmax, __shfl_xor(bmax, 16, 32));
    const float mnew  = fmaxf(mrun, bmax);
    const float alpha = __expf(mrun - mnew);
    float lsum = 0.f;
#pragma unroll
    for (int s = 0; s < 2; ++s)
#pragma unroll
      for (int r = 0; r < 8; ++r) {
        const float p = __expf(pv[s][r] - mnew);
        pv[s][r] = p;
        lsum += p;
      }
    lsum += __shfl_xor(lsum, 16, 32);
    lrun = lrun * alpha + lsum;
    mrun = mnew;
#pragma unroll
    for (int dt = 0; dt < 4; ++dt)
#pragma unroll
      for (int r = 0; r < 8; ++r) oacc[dt][r] *= alpha;

    // ---- P operand (B-layout, 32 keys x 16 queries) via one half-swap ----
    v16bf pop;
#pragma unroll
    for (int j = 0; j < 8; ++j) {
      const float x0 = __shfl_xor(pv[0][j], 16, 32);  // partner's tile0 rows
      const float x1 = __shfl_xor(pv[1][j], 16, 32);  // partner's tile1 rows
      pop[j]     = (__bf16)(half ? x1       : pv[0][j]); // k 0..7  / 16..23
      pop[8 + j] = (__bf16)(half ? pv[1][j] : x0);       // k 8..15 / 24..31
    }

    // ---- O^T update: oacc[dt] += V^T(16d x 32k) * P(32k x 16q) ----
#pragma unroll
    for (int dt = 0; dt < 4; ++dt) {
      const __bf16* vp = Vt + (bh * DH + dt * 16 + l16) * SEQ + kj + half * 16;
      v16bf vop = cat8(*(const v8bf*)vp, *(const v8bf*)(vp + 8));
      oacc[dt] = __builtin_amdgcn_wmma_f32_16x16x32_bf16(false, vop, false, pop,
                                                         (short)0, oacc[dt],
                                                         false, false);
    }
  }

  // ---- normalize and store (f32, [B][T][C]) ----
  const float inv = 1.0f / lrun;
  float* op = O + ((size_t)b * SEQ + qg) * EMB + h * DH;
#pragma unroll
  for (int dt = 0; dt < 4; ++dt)
#pragma unroll
    for (int r = 0; r < 8; ++r)
      op[dt * 16 + (half ? 8 + r : r)] = oacc[dt][r] * inv;
}

extern "C" void kernel_launch(void* const* d_in, const int* in_sizes, int n_in,
                              void* d_out, int out_size, void* d_ws, size_t ws_size,
                              hipStream_t stream) {
  const float*         x   = (const float*)d_in[0];
  const unsigned char* pad = (const unsigned char*)d_in[1];
  const float* wq = (const float*)d_in[2];
  const float* bq = (const float*)d_in[3];
  const float* wk = (const float*)d_in[4];
  const float* bk = (const float*)d_in[5];
  const float* wv = (const float*)d_in[6];
  const float* bv = (const float*)d_in[7];
  const float* wo = (const float*)d_in[8];
  const float* bo = (const float*)d_in[9];
  float* out = (float*)d_out;

  char* ws = (char*)d_ws;
  __bf16* qb = (__bf16*)(ws + ((size_t)0  << 20));   // 16 MB  [B][H][T][DH]
  __bf16* kb = (__bf16*)(ws + ((size_t)16 << 20));   // 16 MB  [B][H][T][DH]
  __bf16* vb = (__bf16*)(ws + ((size_t)32 << 20));   // 16 MB  [B][H][DH][T]
  float*  ob = (float* )(ws + ((size_t)48 << 20));   // 32 MB  [B][T][C]

  const int M = BATCH * SEQ;                         // 8192
  const int gblocks = (M / BM) * (EMB / BN);         // 64*8 = 512
  gemm_bf16_wmma<<<gblocks, 256, 0, stream>>>(x, wq, bq, qb, M, EMB, EMB, 0);
  gemm_bf16_wmma<<<gblocks, 256, 0, stream>>>(x, wk, bk, kb, M, EMB, EMB, 0);
  gemm_bf16_wmma<<<gblocks, 256, 0, stream>>>(x, wv, bv, vb, M, EMB, EMB, 1);
  // waves = B*H*(T/16) = 8192 -> 1024 blocks of 8 waves
  attn_wmma<<<1024, 256, 0, stream>>>(qb, kb, vb, pad, ob);
  gemm_bf16_wmma<<<gblocks, 256, 0, stream>>>(ob, wo, bo, out, M, EMB, EMB, 2);
}
